// SOCA_11450382811433
// MI455X (gfx1250) — matrix operand order repaired and checked by reading, not compiled, not run on hardware
//
#include <hip/hip_runtime.h>
#include <hip/hip_bf16.h>
#include <math.h>

typedef __attribute__((ext_vector_type(16))) __bf16 v16bf;
typedef __attribute__((ext_vector_type(8)))  float  v8f;
typedef __attribute__((ext_vector_type(2)))  float  v2f;

#define C64     64
#define MTOT    65536           // 256*256
#define BATCH   16
#define SLICES  32
#define SLICE_M (MTOT / SLICES) // 2048
#define CHUNK   64              // K elements staged per barrier
#define MINV    (1.0f / 65536.0f)

// ---------------------------------------------------------------------------
// Kernel 1: partial Gram (bf16 WMMA, f32 accum) + partial channel sums.
// One WG = one (batch, M-slice). 256 threads = 8 waves; 16 output tiles of
// 16x16 over the 64x64 Gram; 2 tiles per wave. Double-buffered LDS staging:
// one workgroup barrier per 64-wide K chunk, 4 WMMAs per wave per chunk.
// ---------------------------------------------------------------------------
__global__ void __launch_bounds__(256)
k_gram(const float* __restrict__ x, float* __restrict__ gpart,
       float* __restrict__ spart) {
    __shared__ __bf16 tile[2][C64][CHUNK]; // 16 KB ping-pong bf16 staging
    __shared__ float  red[256];            // rowsum reduction

    const int b    = blockIdx.x / SLICES;
    const int s    = blockIdx.x % SLICES;
    const int tid  = threadIdx.x;
    const int lane = tid & 31;
    const int wave = tid >> 5;

    // cooperative-load assignment: 4 threads per channel, 16 floats each
    const int ch_ld = tid >> 2;
    const int grp   = tid & 3;
    const float* xb = x + (size_t)b * C64 * MTOT;
    const size_t row_base =
        (size_t)ch_ld * MTOT + (size_t)s * SLICE_M + (size_t)grp * 16;

    // tile assignment: wave w owns tiles 2w, 2w+1 (same row-tile ti)
    const int t0   = wave * 2;
    const int ti   = t0 >> 2;
    const int tj0  = t0 & 3;
    const int tj1  = (t0 + 1) & 3;
    const int half = lane >> 4;   // lane group (0: lanes 0-15, 1: lanes 16-31)
    const int lm   = lane & 15;

    v8f acc0, acc1;
#pragma unroll
    for (int r = 0; r < 8; ++r) { acc0[r] = 0.f; acc1[r] = 0.f; }
    float rsum = 0.f;

    for (int kc = 0; kc < SLICE_M / CHUNK; ++kc) {
        const int p = kc & 1;
        // ---- stage 64x64 f32 chunk -> bf16 LDS, accumulate row sums ----
        const float* src = xb + row_base + (size_t)kc * CHUNK;
        float4 f0 = *(const float4*)(src);
        float4 f1 = *(const float4*)(src + 4);
        float4 f2 = *(const float4*)(src + 8);
        float4 f3 = *(const float4*)(src + 12);
        rsum += f0.x + f0.y + f0.z + f0.w + f1.x + f1.y + f1.z + f1.w +
                f2.x + f2.y + f2.z + f2.w + f3.x + f3.y + f3.z + f3.w;
        __bf16* dst = &tile[p][ch_ld][grp * 16];
        dst[0]  = (__bf16)f0.x; dst[1]  = (__bf16)f0.y;
        dst[2]  = (__bf16)f0.z; dst[3]  = (__bf16)f0.w;
        dst[4]  = (__bf16)f1.x; dst[5]  = (__bf16)f1.y;
        dst[6]  = (__bf16)f1.z; dst[7]  = (__bf16)f1.w;
        dst[8]  = (__bf16)f2.x; dst[9]  = (__bf16)f2.y;
        dst[10] = (__bf16)f2.z; dst[11] = (__bf16)f2.w;
        dst[12] = (__bf16)f3.x; dst[13] = (__bf16)f3.y;
        dst[14] = (__bf16)f3.z; dst[15] = (__bf16)f3.w;
        __syncthreads();   // single barrier per chunk (ping-pong buffers)

        // two 32-wide K sub-chunks
#pragma unroll
        for (int ks = 0; ks < CHUNK; ks += 32) {
            // A fragment (16x32): row m = ti*16+lm.
            // comps c -> k = ks + half*16 + c
            v16bf a;
            {
                const __bf16* rp = &tile[p][ti * 16 + lm][ks];
#pragma unroll
                for (int c = 0; c < 8; ++c) a[c]     = rp[half * 16 + c];
#pragma unroll
                for (int c = 0; c < 8; ++c) a[8 + c] = rp[half * 16 + 8 + c];
            }
            // B fragments (32x16): col n = tj*16+lm, comps c -> k = half*16+c
            v16bf b0, b1;
            {
                const __bf16* rp0 = &tile[p][tj0 * 16 + lm][ks + half * 16];
                const __bf16* rp1 = &tile[p][tj1 * 16 + lm][ks + half * 16];
#pragma unroll
                for (int c = 0; c < 16; ++c) { b0[c] = rp0[c]; b1[c] = rp1[c]; }
            }
            acc0 = __builtin_amdgcn_wmma_f32_16x16x32_bf16(
                false, a, false, b0, (short)0, acc0, false, false);
            acc1 = __builtin_amdgcn_wmma_f32_16x16x32_bf16(
                false, a, false, b1, (short)0, acc1, false, false);
        }
    }

    // ---- write partial Gram tiles (C/D layout: m = half*8 + r, n = lm) ----
    float* gp = gpart + (size_t)(b * SLICES + s) * (C64 * C64);
#pragma unroll
    for (int r = 0; r < 8; ++r) {
        int m = half * 8 + r;
        gp[(ti * 16 + m) * C64 + tj0 * 16 + lm] = acc0[r];
        gp[(ti * 16 + m) * C64 + tj1 * 16 + lm] = acc1[r];
    }

    // ---- partial channel sums ----
    red[tid] = rsum;
    __syncthreads();
    if (tid < C64) {
        float v = red[tid * 4] + red[tid * 4 + 1] +
                  red[tid * 4 + 2] + red[tid * 4 + 3];
        spart[(b * SLICES + s) * C64 + tid] = v;
    }
}

// ---------------------------------------------------------------------------
// Kernel 2: reduce partials -> centered covariance  cov = G/M - mu*mu^T
// ---------------------------------------------------------------------------
__global__ void __launch_bounds__(256)
k_reduce_cov(const float* __restrict__ gpart, const float* __restrict__ spart,
             float* __restrict__ cov) {
    const int b = blockIdx.x, tid = threadIdx.x;
    __shared__ float mu[C64];
    if (tid < C64) {
        float s = 0.f;
        for (int sl = 0; sl < SLICES; ++sl)
            s += spart[(b * SLICES + sl) * C64 + tid];
        mu[tid] = s * MINV;
    }
    __syncthreads();
    for (int e = tid; e < C64 * C64; e += 256) {
        float g = 0.f;
        for (int sl = 0; sl < SLICES; ++sl)
            g += gpart[(size_t)(b * SLICES + sl) * (C64 * C64) + e];
        int i = e >> 6, j = e & 63;
        cov[b * (C64 * C64) + e] = g * MINV - mu[i] * mu[j];
    }
}

// ---------------------------------------------------------------------------
// 64x64 f32 matmul across the workgroup using V_WMMA_F32_16X16X4_F32.
// Wave w owns tiles 2w, 2w+1. A frag: comps c -> k = kb + half*2 + c (row lm).
// B frag: comps c -> k = kb + half*2 + c (col lm).
// ---------------------------------------------------------------------------
__device__ __forceinline__ void wg_mm64(const float* __restrict__ A,
                                        const float* __restrict__ Bm,
                                        v8f& acc0, v8f& acc1,
                                        int wave, int lane) {
    const int t0 = wave * 2;
    const int ti = t0 >> 2, tj0 = t0 & 3, tj1 = (t0 + 1) & 3;
    const int half = lane >> 4, lm = lane & 15;
#pragma unroll
    for (int r = 0; r < 8; ++r) { acc0[r] = 0.f; acc1[r] = 0.f; }
#pragma unroll 4
    for (int kb = 0; kb < C64; kb += 4) {
        v2f a, b0, b1;
        const float* ar = &A[(ti * 16 + lm) * C64 + kb + half * 2];
        a[0] = ar[0];
        a[1] = ar[1];
        b0[0] = Bm[(kb + half * 2 + 0) * C64 + tj0 * 16 + lm];
        b0[1] = Bm[(kb + half * 2 + 1) * C64 + tj0 * 16 + lm];
        b1[0] = Bm[(kb + half * 2 + 0) * C64 + tj1 * 16 + lm];
        b1[1] = Bm[(kb + half * 2 + 1) * C64 + tj1 * 16 + lm];
        acc0 = __builtin_amdgcn_wmma_f32_16x16x4_f32(
            false, a, false, b0, (short)0, acc0, false, false);
        acc1 = __builtin_amdgcn_wmma_f32_16x16x4_f32(
            false, a, false, b1, (short)0, acc1, false, false);
    }
}

__device__ __forceinline__ void wg_store64(float* __restrict__ D,
                                           const v8f& acc0, const v8f& acc1,
                                           int wave, int lane, float scale) {
    const int t0 = wave * 2;
    const int ti = t0 >> 2, tj0 = t0 & 3, tj1 = (t0 + 1) & 3;
    const int half = lane >> 4, lm = lane & 15;
#pragma unroll
    for (int r = 0; r < 8; ++r) {
        int m = half * 8 + r;
        D[(ti * 16 + m) * C64 + tj0 * 16 + lm] = acc0[r] * scale;
        D[(ti * 16 + m) * C64 + tj1 * 16 + lm] = acc1[r] * scale;
    }
}

// ---------------------------------------------------------------------------
// Kernel 3: Newton-Schulz matrix sqrt (iterN=5) + row-mean + FC + sigmoid.
// One WG per batch; 3 ping-ponged 64x64 f32 LDS buffers (48 KB).
// ---------------------------------------------------------------------------
__global__ void __launch_bounds__(256)
k_ns(const float* __restrict__ cov, const float* __restrict__ w1,
     const float* __restrict__ b1v, const float* __restrict__ w2,
     const float* __restrict__ b2v, float* __restrict__ att) {
    __shared__ float bufs[3][C64 * C64];
    __shared__ float s_norm;
    __shared__ float vvec[C64];
    __shared__ float hid[8];

    const int b = blockIdx.x, tid = threadIdx.x;
    const int lane = tid & 31, wave = tid >> 5;
    const float* cb = cov + b * (C64 * C64);

    float* Yc = bufs[0];
    float* Zc = bufs[1];
    float* F  = bufs[2];

    // load cov, trace, normalize; Z = 0.5*(3I - A)
    for (int e = tid; e < C64 * C64; e += 256) Yc[e] = cb[e];
    __syncthreads();
    if (tid == 0) {
        float s = 0.f;
        for (int i = 0; i < C64; ++i) s += Yc[i * C64 + i];
        s_norm = s;
    }
    __syncthreads();
    const float normA = s_norm;
    const float inv = 1.0f / normA;
    for (int e = tid; e < C64 * C64; e += 256) {
        float a = Yc[e] * inv;
        Yc[e] = a;                                  // A
        int i = e >> 6, j = e & 63;
        Zc[e] = 0.5f * ((i == j ? 3.0f : 0.0f) - a); // ZY0 = Z1
    }
    __syncthreads();

    v8f a0, a1;
    // Y1 = A @ ZY0 -> F, then rotate roles: Y=F, Z=Zc, free=old A buffer
    wg_mm64(Yc, Zc, a0, a1, wave, lane);
    __syncthreads();
    wg_store64(F, a0, a1, wave, lane, 1.0f);
    __syncthreads();
    { float* t = Yc; Yc = F; F = t; }   // Yc=Y1, Zc=Z1, F=free

    for (int it = 0; it < 3; ++it) {    // range(1, iterN-1)
        // F = 0.5*(3I - Z@Y)
        wg_mm64(Zc, Yc, a0, a1, wave, lane);
        __syncthreads();
        wg_store64(F, a0, a1, wave, lane, 1.0f);
        __syncthreads();
        for (int e = tid; e < C64 * C64; e += 256) {
            int i = e >> 6, j = e & 63;
            F[e] = 0.5f * ((i == j ? 3.0f : 0.0f) - F[e]);
        }
        __syncthreads();
        // newY = Y@F, newZ = F@Z held in registers, then overwrite in place
        v8f y0, y1, z0, z1;
        wg_mm64(Yc, F, y0, y1, wave, lane);
        wg_mm64(F, Zc, z0, z1, wave, lane);
        __syncthreads();
        wg_store64(Yc, y0, y1, wave, lane, 1.0f);
        wg_store64(Zc, z0, z1, wave, lane, 1.0f);
        __syncthreads();
    }

    // final: S = 0.5*sqrt(normA) * Y @ (3I - Z@Y)
    wg_mm64(Zc, Yc, a0, a1, wave, lane);
    __syncthreads();
    wg_store64(F, a0, a1, wave, lane, 1.0f);
    __syncthreads();
    for (int e = tid; e < C64 * C64; e += 256) {
        int i = e >> 6, j = e & 63;
        F[e] = (i == j ? 3.0f : 0.0f) - F[e];
    }
    __syncthreads();
    wg_mm64(Yc, F, a0, a1, wave, lane);
    __syncthreads();
    wg_store64(Zc, a0, a1, wave, lane, 0.5f * sqrtf(normA)); // S -> Zc
    __syncthreads();

    // v[c] = mean over rows; hid = relu(W1 v + b1); att = sigmoid(W2 hid + b2)
    if (tid < C64) {
        float s = 0.f;
        for (int i = 0; i < C64; ++i) s += Zc[i * C64 + tid];
        vvec[tid] = s * (1.0f / 64.0f);
    }
    __syncthreads();
    if (tid < 8) {
        float s = b1v[tid];
        for (int c = 0; c < C64; ++c) s += w1[tid * C64 + c] * vvec[c];
        hid[tid] = s > 0.f ? s : 0.f;
    }
    __syncthreads();
    if (tid < C64) {
        float s = b2v[tid];
        for (int h = 0; h < 8; ++h) s += w2[tid * 8 + h] * hid[h];
        att[b * C64 + tid] = 1.0f / (1.0f + expf(-s));
    }
}

// ---------------------------------------------------------------------------
// Kernel 4: out = att[b,c] * x  (float4 streaming; 16384 float4 per (b,c))
// ---------------------------------------------------------------------------
__global__ void __launch_bounds__(256)
k_scale(const float* __restrict__ x, const float* __restrict__ att,
        float* __restrict__ out) {
    size_t idx = (size_t)blockIdx.x * 256 + threadIdx.x; // float4 index
    int bc = (int)(idx >> 14);                           // MTOT/4 = 16384
    float a = att[bc];
    float4 v = ((const float4*)x)[idx];
    float4 r; r.x = v.x * a; r.y = v.y * a; r.z = v.z * a; r.w = v.w * a;
    ((float4*)out)[idx] = r;
}

// ---------------------------------------------------------------------------
extern "C" void kernel_launch(void* const* d_in, const int* in_sizes, int n_in,
                              void* d_out, int out_size, void* d_ws,
                              size_t ws_size, hipStream_t stream) {
    (void)in_sizes; (void)n_in; (void)out_size; (void)ws_size;
    const float* x  = (const float*)d_in[0];
    const float* w1 = (const float*)d_in[1];
    const float* b1 = (const float*)d_in[2];
    const float* w2 = (const float*)d_in[3];
    const float* b2 = (const float*)d_in[4];
    float* out = (float*)d_out;

    float* ws    = (float*)d_ws;
    float* gpart = ws;                                         // 16*32*4096
    float* spart = gpart + (size_t)BATCH * SLICES * C64 * C64; // 16*32*64
    float* cov   = spart + (size_t)BATCH * SLICES * C64;       // 16*4096
    float* att   = cov + (size_t)BATCH * C64 * C64;            // 16*64

    k_gram<<<dim3(BATCH * SLICES), dim3(256), 0, stream>>>(x, gpart, spart);
    k_reduce_cov<<<dim3(BATCH), dim3(256), 0, stream>>>(gpart, spart, cov);
    k_ns<<<dim3(BATCH), dim3(256), 0, stream>>>(cov, w1, b1, w2, b2, att);
    k_scale<<<dim3(65536), dim3(256), 0, stream>>>(x, att, out);
}
